// FeedBack_73572789780883
// MI455X (gfx1250) — matrix operand order repaired and checked by reading, not compiled, not run on hardware
//
#include <hip/hip_runtime.h>
#include <hip/hip_bf16.h>

// ---------------------------------------------------------------------------
// Problem constants
// ---------------------------------------------------------------------------
#define BATCH     512
#define TSTEPS    1024
#define UNITS     256
#define NFEAT     32
#define OUT_STEPS 32
#define KTOT      288          // 32 (one-hot/pred) + 256 (hidden)
#define NKCH      9            // 288 / 32
#define LDS_KCH   4            // k-chunks permanently cached in LDS
#define WG_BATCH  16           // batch rows per workgroup (== WMMA M)
#define THREADS   512          // 16 wave32 waves
#define A_ST      296          // A-matrix LDS row stride (ushorts), padded
#define ZD_ST     36           // dense-out LDS row stride (floats)
#define WBLK      1024         // bytes per packed (kk, ntile) weight block
#define DENSE_OFF (576 * 1024) // dense_w pack offset in workspace
#define ASZ       (16 * A_ST)  // one A buffer, in ushorts

// dynamic LDS layout (bytes)
#define LW_OFF    0                       // 4 k-chunks x 64 ntiles x 1KB = 262144
#define AB_OFF    262144                  // 2 x 9472
#define BIAS_OFF  281088                  // 4096
#define DB_OFF    285184                  // 128
#define ZD_OFF    285312                  // 2304
#define BASE_OFF  287616                  // 128
#define MU_OFF    287744                  // 64
#define SG_OFF    287808                  // 64
#define BND_OFF   287872                  // 1984
#define IDX_OFF   289856                  // 16384
#define SMEM_BYTES 306240                 // < 320KB WGP LDS limit

typedef __attribute__((ext_vector_type(16))) __bf16       v16bf;
typedef __attribute__((ext_vector_type(8)))  float        v8f;
typedef __attribute__((ext_vector_type(4)))  unsigned int v4u;

union BF16Frag { v16bf v; v4u q[2]; };

__device__ __forceinline__ unsigned short f2bf(float f) {
    unsigned int u = __float_as_uint(f);
    u += 0x7FFFu + ((u >> 16) & 1u);          // round-to-nearest-even
    return (unsigned short)(u >> 16);
}
__device__ __forceinline__ float sigm(float x)      { return 1.0f / (1.0f + __expf(-x)); }
__device__ __forceinline__ float tanh_fast(float x) { return 1.0f - 2.0f / (__expf(2.0f * x) + 1.0f); }

// Acklam inverse-normal CDF + Cornish-Fisher t(df=15) adjustment (fp64, one-time)
__device__ double norm_ppf(double p) {
    const double a0=-39.69683028665376, a1=220.9460984245205, a2=-275.9285104469687,
                 a3=138.357751867269,   a4=-30.66479806614716, a5=2.506628277459239;
    const double b0=-54.47609879822406, b1=161.5858368580409,  b2=-155.6989798598866,
                 b3=66.80131188771972,  b4=-13.28068155288572;
    const double c0=-0.007784894002430293, c1=-0.3223964580411365, c2=-2.400758277161838,
                 c3=-2.549732539343734,    c4=4.374664141464968,   c5=2.938163982698783;
    const double d0=0.007784695709041462, d1=0.3224671290700398,
                 d2=2.445134137142996,    d3=3.754408661907416;
    const double lo = 0.02425, hi = 1.0 - 0.02425;
    if (p < lo) {
        double q = sqrt(-2.0 * log(p));
        return (((((c0*q+c1)*q+c2)*q+c3)*q+c4)*q+c5) / ((((d0*q+d1)*q+d2)*q+d3)*q+1.0);
    }
    if (p > hi) {
        double q = sqrt(-2.0 * log(1.0 - p));
        return -(((((c0*q+c1)*q+c2)*q+c3)*q+c4)*q+c5) / ((((d0*q+d1)*q+d2)*q+d3)*q+1.0);
    }
    double q = p - 0.5, r = q * q;
    return (((((a0*r+a1)*r+a2)*r+a3)*r+a4)*r+a5)*q / (((((b0*r+b1)*r+b2)*r+b3)*r+b4)*r+1.0);
}
__device__ double t_ppf15(double p) {
    const double df = 15.0;
    double z  = norm_ppf(p);
    double z2 = z*z, z3 = z2*z, z5 = z3*z2, z7 = z5*z2, z9 = z7*z2;
    double g1 = (z3 + z) / 4.0;
    double g2 = (5.0*z5 + 16.0*z3 + 3.0*z) / 96.0;
    double g3 = (3.0*z7 + 19.0*z5 + 17.0*z3 - 15.0*z) / 384.0;
    double g4 = (79.0*z9 + 776.0*z7 + 1482.0*z5 - 1920.0*z3 - 945.0*z) / 92160.0;
    return z + g1/df + g2/(df*df) + g3/(df*df*df) + g4/(df*df*df*df);
}

// ---------------------------------------------------------------------------
// One-time weight repack: fp32 -> bf16 in WMMA B-fragment layout.
// Block (kk, nt) = 32 lanes x 32B; lane<16 holds K=kk*32+0..15, lane>=16 K+16,
// VGPR v packs (K=2v, K=2v+1). Combined W' = [kernel(32x1024); rec(256x1024)].
// ---------------------------------------------------------------------------
__global__ void pack_weights(const float* __restrict__ kern,
                             const float* __restrict__ rec,
                             const float* __restrict__ dw,
                             char* __restrict__ ws) {
    int gt  = blockIdx.x * blockDim.x + threadIdx.x;
    int blk = gt >> 5;
    int ln  = gt & 31;
    if (blk < 576) {                            // LSTM weights: 9 kch x 64 ntiles
        int kk = blk >> 6, nt = blk & 63;
        int n  = (nt << 4) + (ln & 15);
        int kb = (kk << 5) + ((ln >> 4) << 4);
        unsigned short* dst = (unsigned short*)(ws + (size_t)blk * WBLK + (ln << 5));
        #pragma unroll
        for (int v = 0; v < 8; ++v) {
            int k0 = kb + 2 * v, k1 = k0 + 1;
            float f0 = (k0 < 32) ? kern[k0 * 1024 + n] : rec[(k0 - 32) * 1024 + n];
            float f1 = (k1 < 32) ? kern[k1 * 1024 + n] : rec[(k1 - 32) * 1024 + n];
            dst[2 * v]     = f2bf(f0);
            dst[2 * v + 1] = f2bf(f1);
        }
    } else if (blk < 592) {                     // dense_w: 8 kch x 2 ntiles
        int b2 = blk - 576;
        int kk = b2 >> 1, nt = b2 & 1;
        int n  = (nt << 4) + (ln & 15);
        int kb = (kk << 5) + ((ln >> 4) << 4);
        unsigned short* dst = (unsigned short*)(ws + DENSE_OFF + (size_t)b2 * WBLK + (ln << 5));
        #pragma unroll
        for (int v = 0; v < 8; ++v) {
            int k0 = kb + 2 * v;
            dst[2 * v]     = f2bf(dw[k0 * 32 + n]);
            dst[2 * v + 1] = f2bf(dw[(k0 + 1) * 32 + n]);
        }
    }
}

// ---------------------------------------------------------------------------
// Persistent LSTM. One WG per 16 batch rows. Gate-interleaved tiling:
// wave w owns columns [w*16, w*16+16) of EVERY gate -> in-register activation,
// cell state lives in a persistent v8f, only h goes through (double-buffered)
// LDS. One barrier per encoder step. Weight stream stays in-loop via an
// opaque (runtime-zero) SGPR offset; loads are plain cached global/ds b128.
// ---------------------------------------------------------------------------
__global__ __launch_bounds__(THREADS, 1) void lstm_persistent(
        const float* __restrict__ inputs, const float* __restrict__ bias,
        const float* __restrict__ dense_b, const char* __restrict__ wpack,
        float* __restrict__ out) {
    extern __shared__ char smem[];
    char*           LW   = smem + LW_OFF;                 // cached weight k-chunks
    unsigned short* Abuf = (unsigned short*)(smem + AB_OFF);
    float*          BIAS = (float*)(smem + BIAS_OFF);
    float*          DB   = (float*)(smem + DB_OFF);
    float*          ZD   = (float*)(smem + ZD_OFF);
    float*          BASE = (float*)(smem + BASE_OFF);
    float*          MU   = (float*)(smem + MU_OFF);
    float*          SG   = (float*)(smem + SG_OFF);
    float*          BND  = (float*)(smem + BND_OFF);
    unsigned char*  IDX  = (unsigned char*)(smem + IDX_OFF);

    const int tid  = threadIdx.x;
    const int w    = tid >> 5;                  // wave id (wave32)
    const int ln   = tid & 31;
    const int b0   = blockIdx.x * WG_BATCH;
    const int arow = ln & 15;                   // A-frag row
    const int aoff = (ln >> 4) << 4;            // A-frag K offset (0/16)
    const int ncol = ln & 15;                   // C/D frag column within tile
    const int m0   = (ln >> 4) << 3;            // C/D frag row base (0/8)
    const int hcol = (w << 4) + ncol;           // this lane's hidden column
    const int lane_blk = (w << 10) + (ln << 5); // wave/lane displacement in a block row

    // ---- setup: quantile table, LDS weight cache, zero state, stage bias ----
    if (tid < 31) {
        double p = 0.0001 + (double)tid * (0.9998 / 30.0);
        BASE[tid] = (float)t_ppf15(p);
    }
    for (int i = tid; i < (LDS_KCH * 64 * WBLK) / 16; i += THREADS)
        ((v4u*)LW)[i] = ((const v4u*)wpack)[i];
    for (int i = tid; i < 2 * ASZ; i += THREADS) Abuf[i] = 0;
    BIAS[tid] = bias[tid]; BIAS[tid + 512] = bias[tid + 512];
    if (tid < 32) DB[tid] = dense_b[tid];
    __syncthreads();

    // ---- mean / std: one wave per batch row, shfl-xor reduction ----
    {
        const float* xr = inputs + (size_t)(b0 + w) * TSTEPS;
        float s = 0.f, s2 = 0.f;
        for (int k = 0; k < 32; ++k) { float v = xr[ln + (k << 5)]; s += v; s2 += v * v; }
        for (int off = 16; off; off >>= 1) {
            s  += __shfl_xor(s,  off, 32);
            s2 += __shfl_xor(s2, off, 32);
        }
        if (ln == 0) {
            float mu  = s * (1.0f / TSTEPS);
            float var = fmaxf(s2 * (1.0f / TSTEPS) - mu * mu, 0.0f);
            float sg  = sqrtf(var);
            MU[w] = mu; SG[w] = sg;
            out[524288 + b0 + w]       = mu;
            out[524288 + 512 + b0 + w] = sg;
        }
    }
    __syncthreads();
    if (tid < 496) { int r = tid / 31, j = tid - r * 31; BND[r * 31 + j] = BASE[j] * SG[r] + MU[r]; }
    __syncthreads();

    // ---- bin index for every (row, t): searchsorted(bounds, x, right) ----
    for (int k = 0; k < 32; ++k) {
        int e = tid + (k << 9);
        int r = e >> 10, t = e & 1023;
        float x = inputs[(size_t)(b0 + r) * TSTEPS + t];
        const float* bb = BND + r * 31;
        int cnt = 0;
        #pragma unroll
        for (int j = 0; j < 31; ++j) cnt += (bb[j] <= x) ? 1 : 0;
        IDX[(r << 10) + t] = (unsigned char)cnt;
    }
    // one-hot for t = 0 into buffer 0
    __syncthreads();
    { int r = tid >> 5, c = tid & 31;
      Abuf[r * A_ST + c] = (IDX[r << 10] == c) ? (unsigned short)0x3F80 : (unsigned short)0; }

    // ---- fused step GEMM: acc[g] = [in|h](16x288) @ Wg(288x16)  (bias later) ----
    // wb/lw bases carry an opaque 0 so LICM cannot hoist the fragment loads.
    auto lstm_gemm = [&](const unsigned short* Ac, const char* wb, const char* lw,
                         v8f acc[4]) {
        const unsigned short* ap0 = Ac + arow * A_ST + aoff;
        #pragma unroll
        for (int kk = 0; kk < NKCH; ++kk) {
            BF16Frag af;
            af.q[0] = *(const v4u*)(ap0 + (kk << 5));
            af.q[1] = *(const v4u*)(ap0 + (kk << 5) + 8);
            BF16Frag bf[4];
            #pragma unroll
            for (int g = 0; g < 4; ++g) {
                const char* p = (kk < LDS_KCH ? lw : wb) + (kk << 16) + (g << 14);
                bf[g].q[0] = *(const v4u*)p;
                bf[g].q[1] = *(const v4u*)(p + 16);
            }
            #pragma unroll
            for (int g = 0; g < 4; ++g) {
                if (kk == 0) {
                    v8f zc = {};                // inline-0 C operand
                    acc[g] = __builtin_amdgcn_wmma_f32_16x16x32_bf16(
                        false, af.v, false, bf[g].v, (short)0, zc, false, false);
                } else {
                    acc[g] = __builtin_amdgcn_wmma_f32_16x16x32_bf16(
                        false, af.v, false, bf[g].v, (short)0, acc[g], false, false);
                }
            }
        }
    };

    float bias_g[4];
    #pragma unroll
    for (int g = 0; g < 4; ++g) bias_g[g] = BIAS[(g << 8) + hcol];
    float db_l[2]; db_l[0] = DB[ncol]; db_l[1] = DB[16 + ncol];

    v8f creg;                                   // persistent cell state (this lane's patch)
    #pragma unroll
    for (int i = 0; i < 8; ++i) creg[i] = 0.0f;

    int cur = 0;
    int opq = 0;                                // opaque 0: defeats LICM, stays an SGPR

    // ================= encoder: 1024 serial steps, 1 barrier each =============
    for (int t = 0; t < TSTEPS; ++t) {
        __syncthreads();
        asm volatile("" : "+s"(opq));           // loop-variant as far as LICM knows
        const char* wb = wpack + opq + lane_blk;
        const char* lw = LW    + opq + lane_blk;
        const unsigned short* Ac = Abuf + cur * ASZ;
        unsigned short*       An = Abuf + (cur ^ 1) * ASZ;

        v8f acc[4];
        lstm_gemm(Ac, wb, lw, acc);

        // in-register LSTM activation; scatter new h (bf16) into next A buffer
        #pragma unroll
        for (int i = 0; i < 8; ++i) {
            float zi = acc[0][i] + bias_g[0];
            float zf = acc[1][i] + bias_g[1];
            float zg = acc[2][i] + bias_g[2];
            float zo = acc[3][i] + bias_g[3];
            float c  = sigm(zf) * creg[i] + sigm(zi) * tanh_fast(zg);
            float h  = sigm(zo) * tanh_fast(c);
            creg[i] = c;
            An[(m0 + i) * A_ST + 32 + hcol] = f2bf(h);
        }
        // one-hot for t+1 into next buffer
        if (t + 1 < TSTEPS) {
            int r = tid >> 5, c = tid & 31;
            An[r * A_ST + c] = (IDX[(r << 10) + t + 1] == c) ? (unsigned short)0x3F80
                                                             : (unsigned short)0;
        }
        cur ^= 1;
    }

    // ================= decoder: dense + softmax feedback =================
    for (int s = 0; s < OUT_STEPS; ++s) {
        __syncthreads();
        asm volatile("" : "+s"(opq));
        const unsigned short* Ac = Abuf + cur * ASZ;
        if (w == 0) {                           // 16x32 dense on wave 0 (2 N-tiles)
            const char* dbse = wpack + DENSE_OFF + opq + (ln << 5);
            v8f dac[2];
            #pragma unroll
            for (int kk = 0; kk < 8; ++kk) {
                BF16Frag af;
                const unsigned short* ap = Ac + arow * A_ST + 32 + (kk << 5) + aoff;
                af.q[0] = *(const v4u*)ap;
                af.q[1] = *(const v4u*)(ap + 8);
                #pragma unroll
                for (int t2 = 0; t2 < 2; ++t2) {
                    const char* p = dbse + (kk << 11) + (t2 << 10);
                    BF16Frag bf;
                    bf.q[0] = *(const v4u*)p;
                    bf.q[1] = *(const v4u*)(p + 16);
                    if (kk == 0) {
                        v8f zc = {};
                        dac[t2] = __builtin_amdgcn_wmma_f32_16x16x32_bf16(
                            false, af.v, false, bf.v, (short)0, zc, false, false);
                    } else {
                        dac[t2] = __builtin_amdgcn_wmma_f32_16x16x32_bf16(
                            false, af.v, false, bf.v, (short)0, dac[t2], false, false);
                    }
                }
            }
            #pragma unroll
            for (int t2 = 0; t2 < 2; ++t2) {
                int n = (t2 << 4) + ncol;
                #pragma unroll
                for (int r = 0; r < 8; ++r)
                    ZD[(m0 + r) * ZD_ST + n] = dac[t2][r] + db_l[t2];
            }
        }
        __syncthreads();
        if (tid < 16) {                         // softmax per row; emit pred + feedback
            int r = tid;
            unsigned short* Aw = Abuf + cur * ASZ;
            float mx = -1e30f;
            for (int c = 0; c < 32; ++c) mx = fmaxf(mx, ZD[r * ZD_ST + c]);
            float sum = 0.f;
            for (int c = 0; c < 32; ++c) sum += __expf(ZD[r * ZD_ST + c] - mx);
            float inv = 1.0f / sum;
            float* po = out + (size_t)(b0 + r) * (OUT_STEPS * NFEAT) + s * NFEAT;
            for (int c = 0; c < 32; ++c) {
                float pv = __expf(ZD[r * ZD_ST + c] - mx) * inv;
                po[c] = pv;
                Aw[r * A_ST + c] = f2bf(pv);
            }
        }
        if (s == OUT_STEPS - 1) break;
        __syncthreads();

        const char* wb = wpack + opq + lane_blk;
        const char* lw = LW    + opq + lane_blk;
        unsigned short* An = Abuf + (cur ^ 1) * ASZ;
        v8f acc[4];
        lstm_gemm(Ac, wb, lw, acc);
        #pragma unroll
        for (int i = 0; i < 8; ++i) {
            float zi = acc[0][i] + bias_g[0];
            float zf = acc[1][i] + bias_g[1];
            float zg = acc[2][i] + bias_g[2];
            float zo = acc[3][i] + bias_g[3];
            float c  = sigm(zf) * creg[i] + sigm(zi) * tanh_fast(zg);
            float h  = sigm(zo) * tanh_fast(c);
            creg[i] = c;
            An[(m0 + i) * A_ST + 32 + hcol] = f2bf(h);
        }
        cur ^= 1;
    }
}

extern "C" void kernel_launch(void* const* d_in, const int* in_sizes, int n_in,
                              void* d_out, int out_size, void* d_ws, size_t ws_size,
                              hipStream_t stream) {
    (void)in_sizes; (void)n_in; (void)out_size; (void)ws_size;
    const float* inputs = (const float*)d_in[0];
    const float* kern   = (const float*)d_in[1];
    const float* rec    = (const float*)d_in[2];
    const float* bias   = (const float*)d_in[3];
    const float* dw     = (const float*)d_in[4];
    const float* db     = (const float*)d_in[5];
    float* out = (float*)d_out;
    char*  ws  = (char*)d_ws;

    pack_weights<<<74, 256, 0, stream>>>(kern, rec, dw, ws);
    lstm_persistent<<<32, THREADS, SMEM_BYTES, stream>>>(inputs, bias, db, ws, out);
}
